// PhasorBlock_15685220565433
// MI455X (gfx1250) — matrix operand order repaired and checked by reading, not compiled, not run on hardware
//
#include <hip/hip_runtime.h>
#include <hip/hip_bf16.h>
#include <math.h>

// ---------------- types ----------------
typedef __attribute__((ext_vector_type(16))) __bf16 v16bf;
typedef __attribute__((ext_vector_type(8)))  float  v8f;

#define PI_F 3.14159265358979323846f

__device__ __forceinline__ v8f wmma_bf16(v16bf a, v16bf b, v8f c) {
  return __builtin_amdgcn_wmma_f32_16x16x32_bf16(false, a, false, b, (short)0, c,
                                                 false, false);
}

// Element (row,k) position inside packed A-operand stream (kchunks = K/32).
__device__ __forceinline__ size_t a_idx(int row, int k, int kchunks) {
  int koff = k & 31;
  int hh = koff >> 4;
  int ii = koff & 15;
  int lane = (row & 15) + 16 * hh;
  return (((size_t)(row >> 4) * kchunks + (k >> 5)) * 32 + lane) * 16 + ii;
}

// Element (r,c) position inside packed B-operand stream (rchnks = R/32).
__device__ __forceinline__ size_t b_idx(int r, int c, int rchnks) {
  int roff = r & 31;
  return (((size_t)(c >> 4) * rchnks + (r >> 5)) * 32 +
          (c & 15) + 16 * (roff >> 4)) * 16 + (roff & 15);
}

// ---------------- pack f32 [R,K] -> A-operand layout -----------------------
__global__ void k_pack_a_f32(const float* __restrict__ s, __bf16* __restrict__ d,
                             int Kk, int n) {
  int o = blockIdx.x * blockDim.x + threadIdx.x;
  if (o >= n) return;
  int i       = o & 15;
  int lane    = (o >> 4) & 31;
  int rest    = o >> 9;
  int kchunks = Kk >> 5;
  int kchunk  = rest % kchunks;
  int mtile   = rest / kchunks;
  int hh  = lane >> 4;
  int koff = 16 * hh + i;
  int row = mtile * 16 + (lane & 15);
  d[o] = (__bf16)s[(size_t)row * Kk + kchunk * 32 + koff];
}

// ---------------- pack f32 [R,C] -> B-operand layout -----------------------
__global__ void k_pack_b_f32(const float* __restrict__ s, __bf16* __restrict__ d,
                             int R, int C, int n) {
  int o = blockIdx.x * blockDim.x + threadIdx.x;
  if (o >= n) return;
  int i      = o & 15;
  int lane   = (o >> 4) & 31;
  int rest   = o >> 9;
  int rchnks = R >> 5;
  int rchunk = rest % rchnks;
  int ctile  = rest / rchnks;
  int row = rchunk * 32 + (lane >> 4) * 16 + i;
  int col = ctile * 16 + (lane & 15);
  d[o] = (__bf16)s[(size_t)row * C + col];
}

// ---------------- WMMA GEMM ------------------------------------------------
// ACT: 0=none, 1=gelu(exact), 2=tanh*pi
// OM:  0=f32 row-major, 1=bf16 row-major, 2=bf16 A-packed, 3=bf16 B-packed
template <int NT, int ACT, int OM>
__global__ __launch_bounds__(32) void k_gemm(
    const __bf16* __restrict__ Ap, const __bf16* __restrict__ Wp,
    const float* __restrict__ bias, void* __restrict__ outp,
    const float* __restrict__ residual, int Nn, int Kk, int Mrows)
{
  int lane = threadIdx.x & 31;
  int half = lane >> 4;
  int l16  = lane & 15;
  int nt0   = blockIdx.x * NT;
  int mtile = blockIdx.y;
  int kchunks = Kk >> 5;

  v8f z = {0.f,0.f,0.f,0.f,0.f,0.f,0.f,0.f};
  v8f acc[NT];
  #pragma unroll
  for (int t = 0; t < NT; ++t) acc[t] = z;

  const __bf16* abase = Ap + ((size_t)mtile * kchunks * 32 + lane) * 16;
  for (int kc = 0; kc < kchunks; ++kc) {
    v16bf a = *(const v16bf*)(abase + (size_t)kc * 512);
    #pragma unroll
    for (int t = 0; t < NT; ++t) {
      const __bf16* bp = Wp + (((size_t)(nt0 + t) * kchunks + kc) * 32 + lane) * 16;
      v16bf b = *(const v16bf*)bp;
      acc[t] = wmma_bf16(a, b, acc[t]);
    }
  }

  int m0 = mtile * 16;
  #pragma unroll
  for (int t = 0; t < NT; ++t) {
    int c = (nt0 + t) * 16 + l16;
    float bv = bias ? bias[c] : 0.f;
    #pragma unroll
    for (int v = 0; v < 8; ++v) {
      int r = m0 + v + 8 * half;
      float val = acc[t][v] + bv;
      if (ACT == 1)      val = 0.5f * val * (1.f + erff(val * 0.70710678118654752f));
      else if (ACT == 2) val = tanhf(val) * PI_F;
      if (residual)      val += residual[(size_t)r * Nn + c];
      if (OM == 0)      ((float*)outp)[(size_t)r * Nn + c] = val;
      else if (OM == 1) ((__bf16*)outp)[(size_t)r * Nn + c] = (__bf16)val;
      else if (OM == 2) ((__bf16*)outp)[a_idx(r, c, Nn >> 5)] = (__bf16)val;
      else              ((__bf16*)outp)[b_idx(r, c, Mrows >> 5)] = (__bf16)val;
    }
  }
}

// ---------------- causal phasor attention ----------------------------------
// Qp: A-packed flat [M,64]; Kx: row-major [*,64]; Vp: B-packed flat [M,D]
// OM: 0 = f32 row-major out; 1 = bf16 A-packed out (flat [M,D])
template <int OM>
__global__ __launch_bounds__(32) void k_attn(
    const __bf16* __restrict__ Qp, const __bf16* __restrict__ Kx,
    const __bf16* __restrict__ Vp, void* __restrict__ outp,
    const float* __restrict__ rowscale, float constscale,
    long kbs, int Ld, int Dd, int Mchunks)
{
  __shared__ __align__(16) __bf16 sS[16 * 32];
  int lane = threadIdx.x & 31;
  int half = lane >> 4;
  int l16  = lane & 15;
  int d0 = blockIdx.x * 64;
  int l0 = blockIdx.y * 16;
  int b  = blockIdx.z;

  const __bf16* Kb = Kx + (size_t)b * kbs;
  int bchunk = b * (Ld >> 5);

  size_t qt = (size_t)((b * Ld + l0) >> 4) * 2;
  v16bf qa0 = *(const v16bf*)(Qp + ((qt + 0) * 32 + lane) * 16);
  v16bf qa1 = *(const v16bf*)(Qp + ((qt + 1) * 32 + lane) * 16);

  v8f z = {0.f,0.f,0.f,0.f,0.f,0.f,0.f,0.f};
  v8f acc[4] = {z, z, z, z};

  for (int j0 = 0; j0 < l0 + 16; j0 += 32) {
    // two 16x16 score tiles; loads always in-bounds (j0 <= l0+15 => jt+15 < L)
    #pragma unroll
    for (int t = 0; t < 2; ++t) {
      int jt = j0 + 16 * t;
      const __bf16* krow = Kb + (size_t)(jt + l16) * 64;
      v16bf kb0 = *(const v16bf*)(krow + 16 * half);
      v16bf kb1 = *(const v16bf*)(krow + 32 + 16 * half);
      v8f s = z;
      s = wmma_bf16(qa0, kb0, s);
      s = wmma_bf16(qa1, kb1, s);
      #pragma unroll
      for (int v = 0; v < 8; ++v) {
        int M  = v + 8 * half;
        int jj = jt + l16;
        float sv = (jj > l0 + M) ? 0.f : s[v];
        int k  = 16 * t + l16;
        int hh = k >> 4;
        int ii = k & 15;
        sS[(M + 16 * hh) * 16 + ii] = (__bf16)sv;
      }
    }
    __syncthreads();
    v16bf sa = *(const v16bf*)(sS + lane * 16);
    __syncthreads();
    int chunk = bchunk + (j0 >> 5);
    #pragma unroll
    for (int t = 0; t < 4; ++t) {
      const __bf16* vp = Vp +
          (((size_t)((d0 >> 4) + t) * Mchunks + chunk) * 32 + lane) * 16;
      v16bf vbt = *(const v16bf*)vp;
      acc[t] = wmma_bf16(sa, vbt, acc[t]);
    }
  }

  #pragma unroll
  for (int t = 0; t < 4; ++t) {
    int cc = d0 + 16 * t + l16;
    #pragma unroll
    for (int v = 0; v < 8; ++v) {
      int rr = l0 + v + 8 * half;
      float sc = constscale * (rowscale ? rowscale[(size_t)b * Ld + rr] : 1.f);
      float val = acc[t][v] * sc;
      if (OM == 0)
        ((float*)outp)[(size_t)b * Ld * Dd + (size_t)rr * Dd + cc] = val;
      else
        ((__bf16*)outp)[a_idx(b * Ld + rr, cc, Dd >> 5)] = (__bf16)val;
    }
  }
}

// ---------------- gate row-dot + sigmoid -----------------------------------
__global__ __launch_bounds__(32) void k_rowdot_sig(
    const __bf16* __restrict__ H, const float* __restrict__ w,
    const float* __restrict__ bsc, float* __restrict__ g, int Kk)
{
  int row = blockIdx.x;
  int lane = threadIdx.x;
  float s = 0.f;
  for (int i = lane; i < Kk; i += 32)
    s += (float)H[(size_t)row * Kk + i] * w[i];
  #pragma unroll
  for (int off = 16; off > 0; off >>= 1) s += __shfl_down(s, off, 32);
  if (lane == 0) g[row] = 1.f / (1.f + expf(-(s + bsc[0])));
}

// ---------------- serial gate scan -----------------------------------------
__global__ void k_scan(const float* __restrict__ vg, const float* __restrict__ kg,
                       float* __restrict__ rs2, int Ld)
{
  int b = blockIdx.x;
  if (threadIdx.x == 0) {
    float a = 0.f;
    for (int l = 0; l < Ld; ++l) {
      a += vg[b * Ld + l];
      float gc = fmaxf(a, 1.f);
      rs2[b * Ld + l] = kg[b * Ld + l] / (sqrtf(gc) * 5.65685424949238f);
    }
  }
}

// ---------------- pass1 phasor Q/K construction ----------------------------
__global__ void k_build_q1(const float* __restrict__ off, const float* __restrict__ pos,
                           __bf16* __restrict__ Q1, __bf16* __restrict__ K1,
                           int Ld, int Pp, int n)
{
  int idx = blockIdx.x * blockDim.x + threadIdx.x;
  if (idx >= n) return;
  int p = idx % Pp;
  int l = (idx / Pp) % Ld;
  int b = idx / (Pp * Ld);
  float ph = pos[l * Pp + p];
  float a = ph + off[idx];
  int row = b * Ld + l;
  Q1[a_idx(row, p, 2)]      = (__bf16)cosf(a);
  Q1[a_idx(row, Pp + p, 2)] = (__bf16)sinf(a);
  if (b == 0) {
    size_t rk = (size_t)l * (2 * Pp);
    K1[rk + p]      = (__bf16)cosf(ph);
    K1[rk + Pp + p] = (__bf16)sinf(ph);
  }
}

// ---------------- phases -> [cos|sin] (A-packed or row-major) --------------
__global__ void k_cossin(const float* __restrict__ ph, __bf16* __restrict__ o,
                         int Pp, int packed, int n)
{
  int idx = blockIdx.x * blockDim.x + threadIdx.x;
  if (idx >= n) return;
  int p = idx % Pp;
  int row = idx / Pp;
  float a = ph[idx];
  if (packed) {
    o[a_idx(row, p, 2)]      = (__bf16)cosf(a);
    o[a_idx(row, Pp + p, 2)] = (__bf16)sinf(a);
  } else {
    o[(size_t)row * (2 * Pp) + p]      = (__bf16)cosf(a);
    o[(size_t)row * (2 * Pp) + Pp + p] = (__bf16)sinf(a);
  }
}

// ---------------- gv = values * value_gate, written B-packed ---------------
__global__ void k_gv_pk(const float* __restrict__ val, const float* __restrict__ vg,
                        __bf16* __restrict__ gv, int Dd, int rchnks, int n)
{
  int o = blockIdx.x * blockDim.x + threadIdx.x;
  if (o >= n) return;
  int i      = o & 15;
  int lane   = (o >> 4) & 31;
  int rest   = o >> 9;
  int rchunk = rest % rchnks;
  int ctile  = rest / rchnks;
  int row = rchunk * 32 + (lane >> 4) * 16 + i;
  int col = ctile * 16 + (lane & 15);
  gv[o] = (__bf16)(val[(size_t)row * Dd + col] * vg[row]);
}

// ---------------- LayerNorm -> A-packed bf16 -------------------------------
__global__ __launch_bounds__(256) void k_ln(
    const float* __restrict__ h, const float* __restrict__ g,
    const float* __restrict__ bb, __bf16* __restrict__ o, int Dd)
{
  int row = blockIdx.x;
  int tid = threadIdx.x;
  __shared__ float red[256];
  const float* hr = h + (size_t)row * Dd;

  float s = 0.f;
  for (int i = tid; i < Dd; i += 256) s += hr[i];
  red[tid] = s; __syncthreads();
  for (int st = 128; st > 0; st >>= 1) {
    if (tid < st) red[tid] += red[tid + st];
    __syncthreads();
  }
  float mean = red[0] / Dd;
  __syncthreads();

  float v = 0.f;
  for (int i = tid; i < Dd; i += 256) { float d0 = hr[i] - mean; v += d0 * d0; }
  red[tid] = v; __syncthreads();
  for (int st = 128; st > 0; st >>= 1) {
    if (tid < st) red[tid] += red[tid + st];
    __syncthreads();
  }
  float inv = rsqrtf(red[0] / Dd + 1e-5f);

  for (int i = tid; i < Dd; i += 256)
    o[a_idx(row, i, Dd >> 5)] = (__bf16)((hr[i] - mean) * inv * g[i] + bb[i]);
}

// =========================== host launcher =================================
extern "C" void kernel_launch(void* const* d_in, const int* in_sizes, int n_in,
                              void* d_out, int out_size, void* d_ws, size_t ws_size,
                              hipStream_t stream)
{
  constexpr int B = 2, L = 1024, D = 512, P = 32;
  constexpr int M = B * L;

  const float* x      = (const float*)d_in[0];
  const float* pos    = (const float*)d_in[1];
  const float* m1_w   = (const float*)d_in[2];
  const float* m1_b   = (const float*)d_in[3];
  const float* off_w1 = (const float*)d_in[4];
  const float* off_b1 = (const float*)d_in[5];
  const float* off_w2 = (const float*)d_in[6];
  const float* off_b2 = (const float*)d_in[7];
  const float* cpe_w  = (const float*)d_in[8];
  const float* cpe_b  = (const float*)d_in[9];
  const float* val_w  = (const float*)d_in[10];
  const float* val_b  = (const float*)d_in[11];
  const float* cg_w1  = (const float*)d_in[12];
  const float* cg_b1  = (const float*)d_in[13];
  const float* cg_w2  = (const float*)d_in[14];
  const float* cg_b2  = (const float*)d_in[15];
  const float* pg_w1  = (const float*)d_in[16];
  const float* pg_b1  = (const float*)d_in[17];
  const float* pg_w2  = (const float*)d_in[18];
  const float* pg_b2  = (const float*)d_in[19];
  const float* ln_g   = (const float*)d_in[20];
  const float* ln_b   = (const float*)d_in[21];
  const float* out_w  = (const float*)d_in[22];
  const float* out_b  = (const float*)d_in[23];
  float* out = (float*)d_out;

  char* base = (char*)d_ws;
  size_t wo = 0;
  auto alloc = [&](size_t bytes) -> void* {
    void* p = base + wo;
    wo += (bytes + 255) & ~(size_t)255;
    return p;
  };

  __bf16* x_pk    = (__bf16*)alloc((size_t)M * D * 2);   // A-packed x
  __bf16* wm1     = (__bf16*)alloc((size_t)D * D * 2);   // B-packed weights
  __bf16* woff1   = (__bf16*)alloc((size_t)D * D * 2);
  __bf16* woff2   = (__bf16*)alloc((size_t)D * P * 2);
  __bf16* wcpe    = (__bf16*)alloc((size_t)D * P * 2);
  __bf16* wval    = (__bf16*)alloc((size_t)D * D * 2);
  __bf16* wcg1    = (__bf16*)alloc((size_t)D * D * 2);
  __bf16* wpg1    = (__bf16*)alloc((size_t)D * D * 2);
  __bf16* wout    = (__bf16*)alloc((size_t)D * D * 2);
  __bf16* c1_pk   = (__bf16*)alloc((size_t)M * D * 2);   // B-packed V (pass1)
  __bf16* hoff_pk = (__bf16*)alloc((size_t)M * D * 2);   // A-packed
  __bf16* hcg_bf  = (__bf16*)alloc((size_t)M * D * 2);   // row-major (row-dot)
  __bf16* hpg_bf  = (__bf16*)alloc((size_t)M * D * 2);
  float*  off_f   = (float*) alloc((size_t)M * P * 4);
  float*  qp_f    = (float*) alloc((size_t)M * P * 4);
  float*  sp_f    = (float*) alloc((size_t)M * P * 4);
  __bf16* Q1      = (__bf16*)alloc((size_t)M * 64 * 2);  // A-packed
  __bf16* K1      = (__bf16*)alloc((size_t)L * 64 * 2);  // row-major
  __bf16* Q2      = (__bf16*)alloc((size_t)M * 64 * 2);  // A-packed
  __bf16* K2      = (__bf16*)alloc((size_t)M * 64 * 2);  // row-major
  float*  val_f   = (float*) alloc((size_t)M * D * 4);
  __bf16* gv_pk   = (__bf16*)alloc((size_t)M * D * 2);   // B-packed V (pass2)
  float*  kg      = (float*) alloc((size_t)M * 4);
  float*  vg      = (float*) alloc((size_t)M * 4);
  float*  rs2     = (float*) alloc((size_t)M * 4);
  __bf16* rc_pk   = (__bf16*)alloc((size_t)M * D * 2);   // A-packed (pass1 out)
  float*  r2_f    = (float*) alloc((size_t)M * D * 4);
  __bf16* ln_pk   = (__bf16*)alloc((size_t)M * D * 2);   // A-packed
  (void)ws_size; (void)n_in; (void)in_sizes; (void)out_size;

  auto packw = [&](const float* s, __bf16* d, int R, int C) {
    k_pack_b_f32<<<(R * C + 255) / 256, 256, 0, stream>>>(s, d, R, C, R * C);
  };

  // ---- stage 0: packed operands straight from f32 ----
  k_pack_a_f32<<<(M * D + 255) / 256, 256, 0, stream>>>(x, x_pk, D, M * D);
  packw(m1_w,   wm1,   D, D);
  packw(off_w1, woff1, D, D);
  packw(off_w2, woff2, D, P);
  packw(cpe_w,  wcpe,  D, P);
  packw(val_w,  wval,  D, D);
  packw(cg_w1,  wcg1,  D, D);
  packw(pg_w1,  wpg1,  D, D);
  packw(out_w,  wout,  D, D);

  dim3 gDD(D / 64, M / 16), gDP(1, M / 16);

  // ---- stage 1: x-driven GEMMs (outputs already in consumer layout) ----
  k_gemm<4,0,3><<<gDD, 32, 0, stream>>>(x_pk, wm1,   m1_b,  c1_pk,   nullptr, D, D, M);
  k_gemm<4,1,2><<<gDD, 32, 0, stream>>>(x_pk, woff1, off_b1, hoff_pk, nullptr, D, D, M);
  k_gemm<2,2,0><<<gDP, 32, 0, stream>>>(hoff_pk, woff2, off_b2, off_f, nullptr, P, D, M);
  k_gemm<2,2,0><<<gDP, 32, 0, stream>>>(x_pk, wcpe,  cpe_b, qp_f,   nullptr, P, D, M);
  k_gemm<4,0,0><<<gDD, 32, 0, stream>>>(x_pk, wval,  val_b, val_f,  nullptr, D, D, M);
  k_gemm<4,1,1><<<gDD, 32, 0, stream>>>(x_pk, wcg1,  cg_b1, hcg_bf, nullptr, D, D, M);
  k_gemm<4,1,1><<<gDD, 32, 0, stream>>>(x_pk, wpg1,  pg_b1, hpg_bf, nullptr, D, D, M);

  // ---- gates ----
  k_rowdot_sig<<<M, 32, 0, stream>>>(hcg_bf, cg_w2, cg_b2, kg, D);
  k_rowdot_sig<<<M, 32, 0, stream>>>(hpg_bf, pg_w2, pg_b2, vg, D);

  // ---- pass 1: causal phasor attention over content -> A-packed rc ----
  k_build_q1<<<(M * P + 255) / 256, 256, 0, stream>>>(off_f, pos, Q1, K1, L, P, M * P);
  k_attn<1><<<dim3(D / 64, L / 16, B), 32, 0, stream>>>(
      Q1, K1, c1_pk, rc_pk, nullptr, 0.17677669529663689f /*1/sqrt(P)*/,
      0L, L, D, M / 32);

  // ---- pass 2 setup ----
  k_gemm<2,2,0><<<gDP, 32, 0, stream>>>(rc_pk, wcpe, cpe_b, sp_f, nullptr, P, D, M);
  k_cossin<<<(M * P + 255) / 256, 256, 0, stream>>>(qp_f, Q2, P, 1, M * P);
  k_cossin<<<(M * P + 255) / 256, 256, 0, stream>>>(sp_f, K2, P, 0, M * P);
  k_gv_pk<<<(M * D + 255) / 256, 256, 0, stream>>>(val_f, vg, gv_pk, D, M / 32, M * D);
  k_scan<<<B, 32, 0, stream>>>(vg, kg, rs2, L);

  // ---- pass 2: causal phasor attention over gated values -> f32 ----
  k_attn<0><<<dim3(D / 64, L / 16, B), 32, 0, stream>>>(
      Q2, K2, gv_pk, r2_f, rs2, 1.0f,
      (long)L * 64, L, D, M / 32);

  // ---- output: x + LN(r2) @ out_w + out_b ----
  k_ln<<<M, 256, 0, stream>>>(r2_f, ln_g, ln_b, ln_pk, D);
  k_gemm<4,0,0><<<gDD, 32, 0, stream>>>(ln_pk, wout, out_b, out, x, D, D, M);
}
